// GraphProcessor_73796128080160
// MI455X (gfx1250) — matrix operand order repaired and checked by reading, not compiled, not run on hardware
//
#include <hip/hip_runtime.h>
#include <hip/hip_bf16.h>

#define N_NODES 50000
#define E_EDGES 600000
#define E_TOT   (E_EDGES + N_NODES)   // 650000 (self loops appended)
#define F_IN    256
#define F_OUT   128
#define G_POOL  8
#define NEG_SLOPE 0.2f

typedef __attribute__((ext_vector_type(16))) __bf16 bf16x16;
typedef __attribute__((ext_vector_type(8)))  float  f32x8;

union FragBF { bf16x16 v; uint4 q[2]; };

__device__ __forceinline__ unsigned short f32_to_bf16(float f) {
    unsigned int u = __float_as_uint(f);
    u = u + 0x7FFFu + ((u >> 16) & 1u);   // round to nearest even
    return (unsigned short)(u >> 16);
}

// float -> order-preserving uint (for atomicMax-based segment max)
__device__ __forceinline__ unsigned int f32_ord(float f) {
    unsigned int u = __float_as_uint(f);
    return (u & 0x80000000u) ? ~u : (u | 0x80000000u);
}
__device__ __forceinline__ float ord_f32(unsigned int u) {
    u = (u & 0x80000000u) ? (u & 0x7FFFFFFFu) : ~u;
    return __uint_as_float(u);
}

// ---------------------------------------------------------------- utilities
__global__ void fill_u32_kernel(unsigned int* __restrict__ p, unsigned int v, size_t n) {
    size_t i = (size_t)blockIdx.x * blockDim.x + threadIdx.x;
    if (i < n) p[i] = v;
}

// 8 floats per thread -> 8 bf16 (one 16B store)
__global__ void convert_bf16_kernel(const float* __restrict__ src,
                                    unsigned short* __restrict__ dst, size_t n) {
    size_t i = ((size_t)blockIdx.x * blockDim.x + threadIdx.x) * 8;
    if (i >= n) return;
    float4 f0 = *(const float4*)(src + i);
    float4 f1 = *(const float4*)(src + i + 4);
    union { unsigned short u[8]; uint4 q; } r;
    r.u[0] = f32_to_bf16(f0.x); r.u[1] = f32_to_bf16(f0.y);
    r.u[2] = f32_to_bf16(f0.z); r.u[3] = f32_to_bf16(f0.w);
    r.u[4] = f32_to_bf16(f1.x); r.u[5] = f32_to_bf16(f1.y);
    r.u[6] = f32_to_bf16(f1.z); r.u[7] = f32_to_bf16(f1.w);
    *(uint4*)(dst + i) = r.q;
}

// W[K][Nout] f32 (row-major) -> Wt[Nout][K] bf16
__global__ void transpose_w_bf16_kernel(const float* __restrict__ W,
                                        unsigned short* __restrict__ Wt,
                                        int K, int Nout) {
    int i = blockIdx.x * blockDim.x + threadIdx.x;
    if (i >= K * Nout) return;
    int k = i / Nout, n = i - k * Nout;
    Wt[(size_t)n * K + k] = f32_to_bf16(W[i]);
}

// ---------------------------------------------------------------- WMMA GEMM
// C[M=50000][128] = A[M][K](bf16) * W[K][128], Bt = W^T stored [128][K] bf16.
// Wt is staged into LDS once per block via async-to-LDS copies (XOR-swizzled
// 16B chunks for bank-conflict-free fragment reads). One wave per 16-row
// stripe, 8 column tiles per wave, K-loop fully unrolled (K is template const).
template <int K>
__global__ void gemm_bf16_wmma_kernel(const unsigned short* __restrict__ A,
                                      const unsigned short* __restrict__ Bt,
                                      float* __restrict__ C) {
    __shared__ unsigned short lds_wt[F_OUT * 256];   // 64KB, K<=256 portion used

    const int tid  = threadIdx.x;
    const int lane = tid & 31;
    const int wave = tid >> 5;

    // ---- cooperative async stage of Wt[128][K] into LDS (swizzled chunks)
    constexpr int CPR    = K / 8;          // 16B chunks per row
    constexpr int CHUNKS = F_OUT * CPR;    // total 16B chunks
#pragma unroll
    for (int c = 0; c < CHUNKS / 256; ++c) {
        int idx  = c * 256 + tid;
        int row  = idx / CPR;
        int ch   = idx - row * CPR;
        int chsw = ch ^ (row & 15);
        unsigned ldsoff = (unsigned)(row * K + chsw * 8) * 2u;  // bytes
        const unsigned short* g = Bt + (size_t)row * K + ch * 8;
        asm volatile("global_load_async_to_lds_b128 %0, %1, off"
                     :: "v"(ldsoff), "v"(g) : "memory");
    }
    asm volatile("s_wait_asynccnt 0x0" ::: "memory");
    __syncthreads();

    const int mtile = blockIdx.x * 8 + wave;          // 16-row tile
    if (mtile * 16 >= N_NODES) return;                // wave-uniform after barrier
    const int mr = lane & 15;                         // A row in tile / C column
    const int hi = lane >> 4;                         // lane-half selector

    const unsigned short* arow = A + (size_t)(mtile * 16 + mr) * K + hi * 8;

    f32x8 acc[8] = {};
#pragma unroll
    for (int kk = 0; kk < K; kk += 32) {
        FragBF a;
        a.q[0] = *(const uint4*)(arow + kk);          // K = kk+h8    .. +8
        a.q[1] = *(const uint4*)(arow + kk + 16);     // K = kk+h8+16 .. +8
        const int c0 = (kk >> 3) + hi;                // first 16B chunk of B frag
        FragBF b[8];
#pragma unroll
        for (int t = 0; t < 8; ++t) {
            const int r = t * 16 + mr;
            const int m = r & 15;
            b[t].q[0] = *(const uint4*)&lds_wt[r * K + ((c0 ^ m) << 3)];
            b[t].q[1] = *(const uint4*)&lds_wt[r * K + (((c0 + 2) ^ m) << 3)];
        }
#pragma unroll
        for (int t = 0; t < 8; ++t) {
            acc[t] = __builtin_amdgcn_wmma_f32_16x16x32_bf16(
                false, a.v, false, b[t].v, (short)0, acc[t], false, false);
        }
    }
#pragma unroll
    for (int t = 0; t < 8; ++t) {
#pragma unroll
        for (int v = 0; v < 8; ++v) {
            int m = mtile * 16 + hi * 8 + v;          // C/D: lanes 0-15 -> M=0..7
            C[(size_t)m * F_OUT + t * 16 + mr] = acc[t][v];
        }
    }
}

// ---------------------------------------------------------------- attention
// wave per node: a_s[n] = h[n]·att_src, a_d[n] = h[n]·att_dst
__global__ void logits_kernel(const float* __restrict__ h,
                              const float* __restrict__ att_s,
                              const float* __restrict__ att_d,
                              float* __restrict__ a_s, float* __restrict__ a_d) {
    int n = blockIdx.x * (blockDim.x >> 5) + (threadIdx.x >> 5);
    int lane = threadIdx.x & 31;
    if (n >= N_NODES) return;
    float4 hv = *(const float4*)(h + (size_t)n * F_OUT + lane * 4);
    float4 s4 = *(const float4*)(att_s + lane * 4);
    float4 d4 = *(const float4*)(att_d + lane * 4);
    float vs = hv.x * s4.x + hv.y * s4.y + hv.z * s4.z + hv.w * s4.w;
    float vd = hv.x * d4.x + hv.y * d4.y + hv.z * d4.z + hv.w * d4.w;
    for (int off = 16; off > 0; off >>= 1) {
        vs += __shfl_down(vs, off, 32);
        vd += __shfl_down(vd, off, 32);
    }
    if (lane == 0) { a_s[n] = vs; a_d[n] = vd; }
}

__device__ __forceinline__ void edge_sd(const int* __restrict__ ei, int e, int& s, int& d) {
    if (e < E_EDGES) { s = ei[e]; d = ei[E_EDGES + e]; }
    else             { s = d = e - E_EDGES; }          // appended self-loops
}

// pass 1: e = leaky_relu(a_s[src]+a_d[dst]); segment max by dst
__global__ void edge_score_max_kernel(const int* __restrict__ ei,
                                      const float* __restrict__ a_s,
                                      const float* __restrict__ a_d,
                                      float* __restrict__ esc,
                                      unsigned int* __restrict__ m_ord) {
    int e = blockIdx.x * blockDim.x + threadIdx.x;
    if (e >= E_TOT) return;
    int s, d; edge_sd(ei, e, s, d);
    float v = a_s[s] + a_d[d];
    v = (v > 0.f) ? v : NEG_SLOPE * v;
    esc[e] = v;
    atomicMax(m_ord + d, f32_ord(v));
}

// pass 2: ex = exp(e - m[dst]); segment sum by dst
__global__ void edge_exp_sum_kernel(const int* __restrict__ ei,
                                    const float* __restrict__ esc,
                                    const unsigned int* __restrict__ m_ord,
                                    float* __restrict__ ex,
                                    float* __restrict__ ssum) {
    int e = blockIdx.x * blockDim.x + threadIdx.x;
    if (e >= E_TOT) return;
    int s, d; edge_sd(ei, e, s, d);
    float t = __expf(esc[e] - ord_f32(m_ord[d]));
    ex[e] = t;
    atomicAdd(ssum + d, t);
}

// pass 3: agg[dst] += (ex/s[dst]) * h[src]   (wave per edge, float4 per lane)
__global__ void edge_aggregate_kernel(const int* __restrict__ ei,
                                      const float* __restrict__ ex,
                                      const float* __restrict__ ssum,
                                      const float* __restrict__ h,
                                      float* __restrict__ agg) {
    int e = blockIdx.x * (blockDim.x >> 5) + (threadIdx.x >> 5);
    int lane = threadIdx.x & 31;
    if (e >= E_TOT) return;
    int s, d; edge_sd(ei, e, s, d);
    float alpha = ex[e] / ssum[d];
    float4 hv = *(const float4*)(h + (size_t)s * F_OUT + lane * 4);
    float* o = agg + (size_t)d * F_OUT + lane * 4;
    atomicAdd(o + 0, alpha * hv.x);
    atomicAdd(o + 1, alpha * hv.y);
    atomicAdd(o + 2, alpha * hv.z);
    atomicAdd(o + 3, alpha * hv.w);
}

__global__ void bias_act_kernel(float* __restrict__ agg, const float* __restrict__ bias,
                                int do_relu) {
    size_t i = (size_t)blockIdx.x * blockDim.x + threadIdx.x;
    if (i >= (size_t)N_NODES * F_OUT) return;
    float v = agg[i] + bias[i & (F_OUT - 1)];
    if (do_relu) v = fmaxf(v, 0.f);
    agg[i] = v;
}

// ---------------------------------------------------------------- pooling
__global__ void pool_accum_kernel(const float* __restrict__ h2,
                                  const int* __restrict__ batch,
                                  float* __restrict__ pool, float* __restrict__ cnt) {
    int n = blockIdx.x * (blockDim.x >> 5) + (threadIdx.x >> 5);
    int lane = threadIdx.x & 31;
    if (n >= N_NODES) return;
    int g = batch[n];
    float4 hv = *(const float4*)(h2 + (size_t)n * F_OUT + lane * 4);
    float* o = pool + (size_t)g * F_OUT + lane * 4;
    atomicAdd(o + 0, hv.x); atomicAdd(o + 1, hv.y);
    atomicAdd(o + 2, hv.z); atomicAdd(o + 3, hv.w);
    if (lane == 0) atomicAdd(cnt + g, 1.0f);
}

__global__ void pool_final_kernel(const float* __restrict__ pool,
                                  const float* __restrict__ cnt,
                                  float* __restrict__ out) {
    int i = blockIdx.x * blockDim.x + threadIdx.x;
    if (i >= G_POOL * F_OUT) return;
    out[i] = pool[i] / fmaxf(cnt[i / F_OUT], 1.0f);
}

// ---------------------------------------------------------------- launch
extern "C" void kernel_launch(void* const* d_in, const int* in_sizes, int n_in,
                              void* d_out, int out_size, void* d_ws, size_t ws_size,
                              hipStream_t stream) {
    const float* x      = (const float*)d_in[0];
    const int*   ei     = (const int*)  d_in[1];
    const int*   batch  = (const int*)  d_in[3];
    const float* W1     = (const float*)d_in[4];
    const float* att_s1 = (const float*)d_in[5];
    const float* att_d1 = (const float*)d_in[6];
    const float* b1     = (const float*)d_in[7];
    const float* W2     = (const float*)d_in[8];
    const float* att_s2 = (const float*)d_in[9];
    const float* att_d2 = (const float*)d_in[10];
    const float* b2     = (const float*)d_in[11];
    float* out = (float*)d_out;

    // workspace carving (256B aligned slices)
    char* ws = (char*)d_ws;
    size_t off = 0;
    auto carve = [&](size_t bytes) -> void* {
        void* p = ws + off;
        off += (bytes + 255) & ~(size_t)255;
        return p;
    };
    unsigned short* xb   = (unsigned short*)carve((size_t)N_NODES * F_IN * 2);
    unsigned short* wt   = (unsigned short*)carve((size_t)F_OUT * F_IN * 2);
    float*          hbuf = (float*)carve((size_t)N_NODES * F_OUT * 4);
    float*          agg  = (float*)carve((size_t)N_NODES * F_OUT * 4);
    float*          a_s  = (float*)carve((size_t)N_NODES * 4);
    float*          a_d  = (float*)carve((size_t)N_NODES * 4);
    unsigned int*   mo   = (unsigned int*)carve((size_t)N_NODES * 4);
    float*          ssum = (float*)carve((size_t)N_NODES * 4);
    float*          esc  = (float*)carve((size_t)E_TOT * 4);
    float*          ex   = (float*)carve((size_t)E_TOT * 4);
    float*          pool = (float*)carve((size_t)G_POOL * F_OUT * 4);
    float*          cnt  = (float*)carve((size_t)G_POOL * 4);

    const int B = 256;
    const int g_conv1 = (int)(((size_t)N_NODES * F_IN / 8 + B - 1) / B);
    const int g_conv2 = (int)(((size_t)N_NODES * F_OUT / 8 + B - 1) / B);
    const int g_gemm  = (N_NODES / 16 + 7) / 8;          // 3125 tiles / 8 waves
    const int g_node  = (N_NODES + 7) / 8;               // wave per node
    const int g_edge  = (E_TOT + B - 1) / B;             // thread per edge
    const int g_eagg  = (E_TOT + 7) / 8;                 // wave per edge
    const int g_nf    = (int)(((size_t)N_NODES * F_OUT + B - 1) / B);
    const int g_n32   = (N_NODES + B - 1) / B;

    for (int layer = 0; layer < 2; ++layer) {
        const int K = layer == 0 ? F_IN : F_OUT;
        if (layer == 0)
            convert_bf16_kernel<<<g_conv1, B, 0, stream>>>(x, xb, (size_t)N_NODES * F_IN);
        else
            convert_bf16_kernel<<<g_conv2, B, 0, stream>>>(agg, xb, (size_t)N_NODES * F_OUT);
        transpose_w_bf16_kernel<<<(K * F_OUT + B - 1) / B, B, 0, stream>>>(
            layer == 0 ? W1 : W2, wt, K, F_OUT);

        if (layer == 0)
            gemm_bf16_wmma_kernel<F_IN ><<<g_gemm, B, 0, stream>>>(xb, wt, hbuf);
        else
            gemm_bf16_wmma_kernel<F_OUT><<<g_gemm, B, 0, stream>>>(xb, wt, hbuf);

        logits_kernel<<<g_node, B, 0, stream>>>(
            hbuf, layer == 0 ? att_s1 : att_s2, layer == 0 ? att_d1 : att_d2, a_s, a_d);

        fill_u32_kernel<<<g_n32, B, 0, stream>>>(mo, 0u, (size_t)N_NODES);
        fill_u32_kernel<<<g_n32, B, 0, stream>>>((unsigned int*)ssum, 0u, (size_t)N_NODES);
        fill_u32_kernel<<<g_nf, B, 0, stream>>>((unsigned int*)agg, 0u, (size_t)N_NODES * F_OUT);

        edge_score_max_kernel<<<g_edge, B, 0, stream>>>(ei, a_s, a_d, esc, mo);
        edge_exp_sum_kernel<<<g_edge, B, 0, stream>>>(ei, esc, mo, ex, ssum);
        edge_aggregate_kernel<<<g_eagg, B, 0, stream>>>(ei, ex, ssum, hbuf, agg);

        bias_act_kernel<<<g_nf, B, 0, stream>>>(agg, layer == 0 ? b1 : b2, layer == 0 ? 1 : 0);
    }

    fill_u32_kernel<<<(G_POOL * F_OUT + B - 1) / B, B, 0, stream>>>(
        (unsigned int*)pool, 0u, (size_t)(G_POOL * F_OUT));
    fill_u32_kernel<<<1, 32, 0, stream>>>((unsigned int*)cnt, 0u, (size_t)G_POOL);
    pool_accum_kernel<<<g_node, B, 0, stream>>>(agg, batch, pool, cnt);
    pool_final_kernel<<<(G_POOL * F_OUT + B - 1) / B, B, 0, stream>>>(pool, cnt, out);
}
